// TextClassifier_11682311045397
// MI455X (gfx1250) — matrix operand order repaired and checked by reading, not compiled, not run on hardware
//
#include <hip/hip_runtime.h>

// ---------------------------------------------------------------------------
// TextClassifier (embedding -> LSTM(S=1024) -> FC -> sigmoid -> BN) for
// MI455X / gfx1250.  Matrix work on v_wmma_f32_16x16x32_bf16 (wave32).
//  - W_hh block slice (64 KB) staged into LDS ONCE for all 1024 steps
//    (async GLOBAL_LOAD_ASYNC_TO_LDS_B128, ASYNCcnt-tracked).
//  - 2-stage software pipeline in the K loops (A on LOADcnt, B on DScnt).
//  - Recurrence is one persistent cooperative kernel with a device-scope
//    generation barrier (avoids ~2 ms of kernel-launch overhead).
// ---------------------------------------------------------------------------

typedef __bf16 bf16_t;
typedef __attribute__((ext_vector_type(16))) __bf16 v16bf;
typedef __attribute__((ext_vector_type(8)))  __bf16 v8bf;
typedef __attribute__((ext_vector_type(8)))  float  v8f;
typedef __attribute__((ext_vector_type(4)))  int    v4i;

// Pointer types expected by the gfx1250 async-to-LDS builtin:
// arg0 = global (AS1) int4*, arg1 = LDS (AS3) int4*.
typedef __attribute__((address_space(1))) v4i as1_v4i;
typedef __attribute__((address_space(3))) v4i as3_v4i;

#define BB   128        // batch
#define SS   1024       // seq len
#define DD   512        // embed dim
#define HH   512        // hidden
#define GG   2048       // 4*H
#define ROWS (SS * BB)  // 131072
#define LSTM_BLOCKS 32

// ---- helpers --------------------------------------------------------------

__device__ __forceinline__ bf16_t f2bf(float f) {
  unsigned u = __float_as_uint(f);
  u += 0x7FFFu + ((u >> 16) & 1u);          // round-to-nearest-even
  unsigned short s = (unsigned short)(u >> 16);
  return __builtin_bit_cast(bf16_t, s);
}

__device__ __forceinline__ float sigmoidf(float x) {
  return 1.0f / (1.0f + __expf(-x));
}
__device__ __forceinline__ float tanh_fast(float x) {
  return 1.0f - 2.0f / (__expf(2.0f * x) + 1.0f);
}

// A-fragment (16x32 bf16, row-major source).
// Lanes 0-15: row M=lane, K {0..15}; lanes 16-31: K {16..31}.
__device__ __forceinline__ v16bf load_a_frag(const bf16_t* __restrict__ rowp,
                                             int k0, int half) {
  const v8bf lo = *(const v8bf*)(rowp + k0 + half * 16);
  const v8bf hi = *(const v8bf*)(rowp + k0 + half * 16 + 8);
  return __builtin_shufflevector(lo, hi, 0, 1, 2, 3, 4, 5, 6, 7,
                                         8, 9, 10, 11, 12, 13, 14, 15);
}

// Stage nelem bf16 from global to LDS, block-cooperatively, 16B chunks.
// Prefers the CDNA5 async-to-LDS path (ASYNCcnt); falls back to ds_store.
__device__ __forceinline__ void stage_to_lds(bf16_t* dst_lds,
                                             const bf16_t* src_g, int nelem,
                                             int tid, int nthreads) {
#if __has_builtin(__builtin_amdgcn_global_load_async_to_lds_b128)
  for (int i = tid; i < nelem / 8; i += nthreads) {
    __builtin_amdgcn_global_load_async_to_lds_b128(
        (as1_v4i*)(src_g + (size_t)i * 8),
        (as3_v4i*)(dst_lds + (size_t)i * 8),
        /*offset=*/0, /*cpol=*/0);
  }
#if __has_builtin(__builtin_amdgcn_s_wait_asynccnt)
  __builtin_amdgcn_s_wait_asynccnt(0);
#else
  asm volatile("s_wait_asynccnt 0x0" ::: "memory");
#endif
#else
  for (int i = tid; i < nelem / 8; i += nthreads)
    *(v8bf*)(dst_lds + (size_t)i * 8) = *(const v8bf*)(src_g + (size_t)i * 8);
#endif
}

// Device-scope generation barrier across all blocks of a cooperative grid.
__device__ __forceinline__ void grid_barrier(unsigned* cnt, unsigned* gen,
                                             unsigned nblocks) {
  __syncthreads();
  if (threadIdx.x == 0) {
    unsigned g = __hip_atomic_load(gen, __ATOMIC_ACQUIRE,
                                   __HIP_MEMORY_SCOPE_AGENT);
    unsigned prev = __hip_atomic_fetch_add(cnt, 1u, __ATOMIC_ACQ_REL,
                                           __HIP_MEMORY_SCOPE_AGENT);
    if (prev == nblocks - 1) {
      __hip_atomic_store(cnt, 0u, __ATOMIC_RELAXED, __HIP_MEMORY_SCOPE_AGENT);
      __hip_atomic_fetch_add(gen, 1u, __ATOMIC_ACQ_REL,
                             __HIP_MEMORY_SCOPE_AGENT);
    } else {
      while (__hip_atomic_load(gen, __ATOMIC_ACQUIRE,
                               __HIP_MEMORY_SCOPE_AGENT) == g) {
        __builtin_amdgcn_s_sleep(1);
      }
    }
  }
  __syncthreads();
}

// ---- 0. init: state buffers + barrier counters (re-run every launch) ------

__global__ void __launch_bounds__(256)
init_kernel(const float* __restrict__ h0, const float* __restrict__ c0,
            bf16_t* __restrict__ h_bf, float* __restrict__ c,
            unsigned* __restrict__ bar) {
  int idx = blockIdx.x * 256 + threadIdx.x;      // 65536 = B*H
  if (idx < 2) bar[idx] = 0u;
  h_bf[idx] = f2bf(h0[idx]);
  c[idx] = c0[idx];
}

// ---- 1. fp32 -> bf16 weight conversion ------------------------------------

__global__ void __launch_bounds__(256)
cvt_bf16_kernel(const float* __restrict__ src, bf16_t* __restrict__ dst,
                int n) {
  for (int i = blockIdx.x * 256 + threadIdx.x; i < n; i += gridDim.x * 256)
    dst[i] = f2bf(src[i]);
}

// ---- 2. embedding gather, time-major, bf16: X[s*B+b][:] = emb[tok[b][s]] --

__global__ void __launch_bounds__(256)
embed_kernel(const int* __restrict__ tokens, const float* __restrict__ emb,
             bf16_t* __restrict__ X) {
  const int lane = threadIdx.x & 31;
  const int row  = (blockIdx.x * 256 + threadIdx.x) >> 5;  // s*B + b
  const int s = row >> 7;            // row / 128
  const int b = row & 127;
  const int tok = tokens[b * SS + s];
  const float4* __restrict__ src = (const float4*)(emb + (size_t)tok * DD);
  bf16_t* __restrict__ dst = X + (size_t)row * DD;
#pragma unroll
  for (int it = 0; it < 4; ++it) {
    float4 v = src[it * 32 + lane];
    int o = (it * 32 + lane) * 4;
    dst[o + 0] = f2bf(v.x);
    dst[o + 1] = f2bf(v.y);
    dst[o + 2] = f2bf(v.z);
    dst[o + 3] = f2bf(v.w);
  }
}

// ---- 3. XG[r][g] = sum_k X[r][k] * Wih[g][k] + b_ih[g] + b_hh[g] ----------
// Block: 8 waves share one 16x512 X tile (staged in LDS); each wave owns a
// 16x64 output strip (4 N-tiles), B (W_ih) streamed from L2, K software-
// pipelined 2 stages deep.

__global__ void __launch_bounds__(256)
gemm_xg_kernel(const bf16_t* __restrict__ X, const bf16_t* __restrict__ Wih,
               const float* __restrict__ b_ih, const float* __restrict__ b_hh,
               float* __restrict__ XG) {
  __shared__ bf16_t Ash[16 * DD];                     // 16 KB
  const int lane = threadIdx.x & 31;
  const int wave = threadIdx.x >> 5;
  const int m0 = (blockIdx.x >> 2) * 16;              // 8192 M-tiles
  const int n0 = (((blockIdx.x & 3) << 3) + wave) * 64;  // 32 N-groups
  const int half = lane >> 4;
  const int r = lane & 15;

  stage_to_lds(Ash, X + (size_t)m0 * DD, 16 * DD, threadIdx.x, 256);
  __syncthreads();

  const bf16_t* __restrict__ pa = Ash + (size_t)r * DD;          // LDS
  const bf16_t* __restrict__ pb = Wih + (size_t)(n0 + r) * DD + half * 16;

  v8f acc[4] = {v8f{}, v8f{}, v8f{}, v8f{}};
  v16bf a_c = load_a_frag(pa, 0, half);
  v16bf b_c[4];
#pragma unroll
  for (int t = 0; t < 4; ++t)
    b_c[t] = *(const v16bf*)(pb + (size_t)t * 16 * DD);

#pragma unroll
  for (int k0 = 0; k0 < DD; k0 += 32) {
    v16bf a_n = a_c;
    v16bf b_n[4] = {b_c[0], b_c[1], b_c[2], b_c[3]};
    if (k0 + 32 < DD) {
      a_n = load_a_frag(pa, k0 + 32, half);
#pragma unroll
      for (int t = 0; t < 4; ++t)
        b_n[t] = *(const v16bf*)(pb + (size_t)t * 16 * DD + k0 + 32);
    }
#pragma unroll
    for (int t = 0; t < 4; ++t)
      acc[t] = __builtin_amdgcn_wmma_f32_16x16x32_bf16(
          false, a_c, false, b_c[t], (short)0, acc[t], false, false);
    a_c = a_n;
#pragma unroll
    for (int t = 0; t < 4; ++t) b_c[t] = b_n[t];
  }

#pragma unroll
  for (int v = 0; v < 8; ++v) {
    const int row = m0 + half * 8 + v;
    float* __restrict__ o = XG + (size_t)row * GG;
#pragma unroll
    for (int t = 0; t < 4; ++t) {
      const int cN = n0 + t * 16 + r;
      o[cN] = acc[t][v] + b_ih[cN] + b_hh[cN];
    }
  }
}

// ---- 4. persistent LSTM recurrence ----------------------------------------
// 32 blocks x 256 threads.  Block b owns output columns [64b, 64b+64); its
// 64 W_hh rows (64 KB) are staged into LDS ONCE and reused for all 1024
// steps.  Per step: phase 1 gates = XG[s] + h @ Whh^T (wave w = M-tile w,
// 4 N-tiles, A from global h_bf / B from LDS, 2-stage pipeline); phase 2
// elementwise c/h update.  Device-scope barrier between phases.

__global__ void __launch_bounds__(256)
lstm_kernel(const float* __restrict__ XG, const bf16_t* __restrict__ Whh,
            bf16_t* __restrict__ h_bf, float* __restrict__ c,
            float* __restrict__ h_out, float* __restrict__ gates,
            unsigned* __restrict__ bar) {
  __shared__ bf16_t Bsh[64 * HH];                 // 64 KB, lives whole kernel
  const int lane = threadIdx.x & 31;
  const int wave = threadIdx.x >> 5;
  const int m0 = wave * 16;                       // 8 M-tiles over 8 waves
  const int n0 = blockIdx.x * 64;                 // block's N range
  const int half = lane >> 4;
  const int r = lane & 15;
  const int gtid = blockIdx.x * 256 + threadIdx.x;  // 0..8191

  stage_to_lds(Bsh, Whh + (size_t)n0 * HH, 64 * HH, threadIdx.x, 256);
  __syncthreads();

  const bf16_t* __restrict__ pbl = Bsh + (size_t)r * HH + half * 16;  // LDS

  for (int s = 0; s < SS; ++s) {
    const float* __restrict__ xgs = XG + (size_t)s * (BB * GG);
    if (s + 1 < SS)
      __builtin_prefetch(xgs + (size_t)(BB * GG) + (size_t)(m0 + r) * GG + n0,
                         0, 0);

    v8f acc[4];
#pragma unroll
    for (int t = 0; t < 4; ++t)
#pragma unroll
      for (int v = 0; v < 8; ++v)
        acc[t][v] = xgs[(size_t)(m0 + half * 8 + v) * GG + (n0 + t * 16 + r)];

    const bf16_t* __restrict__ pa = h_bf + (size_t)(m0 + r) * HH;  // global
    v16bf a_c = load_a_frag(pa, 0, half);
    v16bf b_c[4];
#pragma unroll
    for (int t = 0; t < 4; ++t)
      b_c[t] = *(const v16bf*)(pbl + (size_t)t * 16 * HH);

#pragma unroll
    for (int k0 = 0; k0 < HH; k0 += 32) {
      v16bf a_n = a_c;
      v16bf b_n[4] = {b_c[0], b_c[1], b_c[2], b_c[3]};
      if (k0 + 32 < HH) {
        a_n = load_a_frag(pa, k0 + 32, half);
#pragma unroll
        for (int t = 0; t < 4; ++t)
          b_n[t] = *(const v16bf*)(pbl + (size_t)t * 16 * HH + k0 + 32);
      }
#pragma unroll
      for (int t = 0; t < 4; ++t)
        acc[t] = __builtin_amdgcn_wmma_f32_16x16x32_bf16(
            false, a_c, false, b_c[t], (short)0, acc[t], false, false);
      a_c = a_n;
#pragma unroll
      for (int t = 0; t < 4; ++t) b_c[t] = b_n[t];
    }

#pragma unroll
    for (int t = 0; t < 4; ++t)
#pragma unroll
      for (int v = 0; v < 8; ++v)
        gates[(size_t)(m0 + half * 8 + v) * GG + (n0 + t * 16 + r)] =
            acc[t][v];

    grid_barrier(bar, bar + 1, LSTM_BLOCKS);

    // elementwise state update: B*H = 65536 elems, 8192 threads, stride 8192
#pragma unroll
    for (int e = 0; e < 8; ++e) {
      const int idx = e * 8192 + gtid;
      const int b  = idx >> 9;          // idx / 512
      const int j  = idx & 511;
      const float* __restrict__ gr = gates + (size_t)b * GG;
      const float iv = sigmoidf(gr[j]);
      const float fv = sigmoidf(gr[512 + j]);
      const float gv = tanh_fast(gr[1024 + j]);
      const float ov = sigmoidf(gr[1536 + j]);
      const float cv = fv * c[idx] + iv * gv;
      const float hv = ov * tanh_fast(cv);
      c[idx] = cv;
      h_bf[idx] = f2bf(hv);
      if (s == SS - 1) h_out[idx] = hv;
    }

    grid_barrier(bar, bar + 1, LSTM_BLOCKS);
  }
}

// ---- 5. head: y = BN(sigmoid(h @ Wfc^T + b_fc)) ---------------------------

__global__ void __launch_bounds__(256)
head_kernel(const float* __restrict__ h, const float* __restrict__ Wfc,
            const float* __restrict__ bfc, const float* __restrict__ gamma,
            const float* __restrict__ beta, const float* __restrict__ mean,
            const float* __restrict__ var, float* __restrict__ out) {
  const int t = threadIdx.x;        // 256 = 128 * 2
  const int b = t >> 1;
  const int j = t & 1;
  const float4* __restrict__ hr = (const float4*)(h + (size_t)b * HH);
  const float4* __restrict__ wr = (const float4*)(Wfc + (size_t)j * HH);
  float acc = 0.f;
#pragma unroll 8
  for (int k = 0; k < HH / 4; ++k) {
    float4 a = hr[k], w = wr[k];
    acc += a.x * w.x + a.y * w.y + a.z * w.z + a.w * w.w;
  }
  const float y = sigmoidf(acc + bfc[j]);
  out[t] = gamma[j] * (y - mean[j]) * __frsqrt_rn(var[j] + 1e-5f) + beta[j];
}

// ---------------------------------------------------------------------------

extern "C" void kernel_launch(void* const* d_in, const int* in_sizes, int n_in,
                              void* d_out, int out_size, void* d_ws,
                              size_t ws_size, hipStream_t stream) {
  (void)in_sizes; (void)n_in; (void)out_size; (void)ws_size;
  const int*   tokens = (const int*)  d_in[0];
  const float* emb    = (const float*)d_in[1];
  const float* W_ih   = (const float*)d_in[2];
  const float* W_hh   = (const float*)d_in[3];
  const float* b_ih   = (const float*)d_in[4];
  const float* b_hh   = (const float*)d_in[5];
  const float* h0     = (const float*)d_in[6];
  const float* c0     = (const float*)d_in[7];
  const float* W_fc   = (const float*)d_in[8];
  const float* b_fc   = (const float*)d_in[9];
  const float* gamma  = (const float*)d_in[10];
  const float* beta   = (const float*)d_in[11];
  const float* mean   = (const float*)d_in[12];
  const float* var    = (const float*)d_in[13];
  float* out = (float*)d_out;

  // workspace carve (~1.22 GB): Xbf 128MB, weights 4MB, XG 1GiB, state ~2MB
  char* ws = (char*)d_ws;
  size_t off = 0;
  auto carve = [&](size_t bytes) -> void* {
    void* p = ws + off;
    off += (bytes + 255) & ~(size_t)255;
    return p;
  };
  bf16_t*   Xbf   = (bf16_t*)  carve((size_t)ROWS * DD * 2);
  bf16_t*   Wihb  = (bf16_t*)  carve((size_t)GG * DD * 2);
  bf16_t*   Whhb  = (bf16_t*)  carve((size_t)GG * HH * 2);
  float*    XG    = (float*)   carve((size_t)ROWS * GG * 4);
  float*    gates = (float*)   carve((size_t)BB * GG * 4);
  bf16_t*   hbf   = (bf16_t*)  carve((size_t)BB * HH * 2);
  float*    hf    = (float*)   carve((size_t)BB * HH * 4);
  float*    cbuf  = (float*)   carve((size_t)BB * HH * 4);
  unsigned* bar   = (unsigned*)carve(256);

  init_kernel<<<BB * HH / 256, 256, 0, stream>>>(h0, c0, hbf, cbuf, bar);
  cvt_bf16_kernel<<<1024, 256, 0, stream>>>(W_ih, Wihb, GG * DD);
  cvt_bf16_kernel<<<1024, 256, 0, stream>>>(W_hh, Whhb, GG * HH);
  embed_kernel<<<ROWS / 8, 256, 0, stream>>>(tokens, emb, Xbf);
  gemm_xg_kernel<<<(ROWS / 16) * 4, 256, 0, stream>>>(Xbf, Wihb, b_ih, b_hh,
                                                      XG);
  lstm_kernel<<<LSTM_BLOCKS, 256, 0, stream>>>(XG, Whhb, hbf, cbuf, hf,
                                               gates, bar);
  head_kernel<<<1, 256, 0, stream>>>(hf, W_fc, b_fc, gamma, beta, mean, var,
                                     out);
}
